// myConv2d_11828339933651
// MI455X (gfx1250) — matrix operand order repaired and checked by reading, not compiled, not run on hardware
//
#include <hip/hip_runtime.h>

typedef __attribute__((ext_vector_type(16))) __bf16 v16bf;
typedef __attribute__((ext_vector_type(8)))  __bf16 v8bf;
typedef __attribute__((ext_vector_type(8)))  float  v8f;
typedef __attribute__((ext_vector_type(4)))  unsigned int v4u;

#define KP  40                    // padded K pitch (bf16): 80B rows -> 16B-aligned b128 frags
#define OPD 65                    // padded pixel pitch (f32) for epilogue staging
#define A_PLANE_B (64 * KP * 2)   // 5120 B
#define B_PLANE_B (128 * KP * 2)  // 10240 B
#define BUF_BYTES (2 * A_PLANE_B + 2 * B_PLANE_B)  // 30720 B per pipeline buffer

__device__ __forceinline__ v16bf cat8(v8bf a, v8bf b) {
    return __builtin_shufflevector(a, b, 0,1,2,3,4,5,6,7,8,9,10,11,12,13,14,15);
}

// One-time weight pre-conversion: f32 OIHW -> bf16 hi/lo planes, layout
// wbf[((rs*2 + plane)*256 + n)*128 + c]   (c contiguous -> b128 loads in main loop)
__global__ __launch_bounds__(256)
void preconv_weights(const float* __restrict__ w, __bf16* __restrict__ wbf) {
    int flat = blockIdx.x * 256 + threadIdx.x;     // 0 .. 294911
    int c  = flat & 127;
    int n  = (flat >> 7) & 255;
    int rs = flat >> 15;                           // 0..8
    float v = w[(size_t)(n * 128 + c) * 9 + rs];
    __bf16 vh = (__bf16)v;
    wbf[((size_t)(rs * 2 + 0) * 256 + n) * 128 + c] = vh;
    wbf[((size_t)(rs * 2 + 1) * 256 + n) * 128 + c] = (__bf16)(v - (float)vh);
}

template <bool PRE>
__global__ __launch_bounds__(256)
void conv3x3_igemm_bf16x3(const float* __restrict__ x,
                          const float* __restrict__ w,
                          const __bf16* __restrict__ wbf,
                          const float* __restrict__ bias,
                          float* __restrict__ out)
{
    __shared__ __align__(16) unsigned char smem[2 * BUF_BYTES];  // 61440 B

    const int t    = threadIdx.x;
    const int lane = t & 31;
    const int wv   = t >> 5;                  // wave 0..7
    const int wm   = (wv & 1) * 32;           // wave M offset (pixels)
    const int wn   = (wv >> 1) * 32;          // wave N offset (out channels)

    const int n_img  = blockIdx.x / 49;
    const int p_base = (blockIdx.x % 49) * 64;   // 3136 = 49*64
    const int n_base = blockIdx.y * 128;

    const float* xN = x + (size_t)n_img * 128 * 3136;

    // ---- loop-invariant per-thread offsets & pad-validity masks ----------------
    // addr(x) = xN + [cb*3136 + (r-1)*56 + (s-1)]uniform + [cl*3136 + pix]invariant
    int offA[8];      // element offsets into x (per A-staging slot)
    int mskA[8];      // bits 0..2: h+r-1 in range; bits 3..5: w+s-1 in range
    int offLA[8];     // LDS element offsets [pl][cl]
#pragma unroll
    for (int i = 0; i < 8; ++i) {
        int idx = i * 256 + t;
        int pl  = idx & 63;
        int cl  = idx >> 6;
        int pix = p_base + pl;
        int h   = pix / 56;
        int w_  = pix % 56;
        offA[i]  = cl * 3136 + pix;          // h*56 + w == pix
        offLA[i] = pl * KP + cl;
        int m = 0;
#pragma unroll
        for (int r = 0; r < 3; ++r) if ((unsigned)(h  + r - 1) < 56u) m |= 1 << r;
#pragma unroll
        for (int s = 0; s < 3; ++s) if ((unsigned)(w_ + s - 1) < 56u) m |= 8 << s;
        mskA[i] = m;
    }
    int offB[4], offLB[4];                   // PRE path (b128 bf16 chunks)
#pragma unroll
    for (int i = 0; i < 4; ++i) {
        int plane = i >> 1;
        int rem   = ((i & 1) << 8) + t;      // 0..511
        int nl    = rem >> 2;
        int c8    = (rem & 3) << 3;
        offB[i]  = (plane * 256 + n_base + nl) * 128 + c8;
        offLB[i] = nl * KP + c8;
    }
    int offW[16], offLW[16];                 // fallback path (f32 weights)
    if (!PRE) {
#pragma unroll
        for (int i = 0; i < 16; ++i) {
            int idx = i * 256 + t;
            int cl = idx & 31, nl = idx >> 5;
            offW[i]  = ((n_base + nl) * 128 + cl) * 9;
            offLW[i] = nl * KP + cl;
        }
    }

    v8f acc[2][2] = {};

    // ---------------- staging helpers (slice k: rs = k>>2 tap, cb = (k&3)*32) ----
    auto loadA = [&](int k, float* gA) {
        int rs = k >> 2, cb = (k & 3) << 5;
        int r = rs / 3, s = rs - 3 * r;
        const float* xs = xN + (cb * 3136 + (r - 1) * 56 + (s - 1));  // uniform shift
#pragma unroll
        for (int i = 0; i < 8; ++i) {
            bool ok = ((mskA[i] >> r) & (mskA[i] >> (3 + s)) & 1) != 0;
            int  o  = ok ? offA[i] : 64;     // safe in-buffer offset when padding
            float v = xs[o];
            gA[i] = ok ? v : 0.0f;
        }
    };
    auto storeA = [&](const float* gA, unsigned char* base) {
        __bf16* sAh = (__bf16*)base;
        __bf16* sAl = (__bf16*)(base + A_PLANE_B);
#pragma unroll
        for (int i = 0; i < 8; ++i) {
            __bf16 vh = (__bf16)gA[i];
            sAh[offLA[i]] = vh;
            sAl[offLA[i]] = (__bf16)(gA[i] - (float)vh);
        }
    };
    auto loadB = [&](int k, v4u* gB) {       // pure bf16 b128 copy, no VALU
        int rs = k >> 2, cb = (k & 3) << 5;
        const __bf16* ws_ = wbf + rs * (2 * 256 * 128) + cb;   // uniform shift
#pragma unroll
        for (int i = 0; i < 4; ++i)
            gB[i] = *(const v4u*)&ws_[offB[i]];
    };
    auto storeB = [&](const v4u* gB, unsigned char* base) {
#pragma unroll
        for (int i = 0; i < 4; ++i) {
            int plane = i >> 1;
            __bf16* dst = (__bf16*)(base + 2 * A_PLANE_B + plane * B_PLANE_B);
            *(v4u*)&dst[offLB[i]] = gB[i];
        }
    };
    auto loadBf = [&](int k, float* gBf) {   // fallback: f32 weights
        int rs = k >> 2, cb = (k & 3) << 5;
        const float* ws_ = w + cb * 9 + rs;  // uniform shift
#pragma unroll
        for (int i = 0; i < 16; ++i)
            gBf[i] = ws_[offW[i]];
    };
    auto storeBf = [&](const float* gBf, unsigned char* base) {
        __bf16* bh = (__bf16*)(base + 2 * A_PLANE_B);
        __bf16* bl = (__bf16*)(base + 2 * A_PLANE_B + B_PLANE_B);
#pragma unroll
        for (int i = 0; i < 16; ++i) {
            __bf16 vh = (__bf16)gBf[i];
            bh[offLW[i]] = vh;
            bl[offLW[i]] = (__bf16)(gBf[i] - (float)vh);
        }
    };

    // ---------------- WMMA block: 12 x v_wmma per wave per slice ------------------
    auto compute = [&](const unsigned char* base) {
        const __bf16* sAh = (const __bf16*)base;
        const __bf16* sAl = (const __bf16*)(base + A_PLANE_B);
        const __bf16* sBh = (const __bf16*)(base + 2 * A_PLANE_B);
        const __bf16* sBl = (const __bf16*)(base + 2 * A_PLANE_B + B_PLANE_B);

        const int am  = lane & 15;
        const int ak0 = (lane >> 4) * 8;    // A: K runs {k0..k0+7},{k0+16..k0+23}
        const int bk0 = (lane >> 4) * 8;    // B: K runs {k0..k0+7},{k0+16..k0+23}

        v16bf aHi[2], aLo[2], bHi[2], bLo[2];
#pragma unroll
        for (int mi = 0; mi < 2; ++mi) {
            int m = wm + mi * 16 + am;
            aHi[mi] = cat8(*(const v8bf*)&sAh[m * KP + ak0],
                           *(const v8bf*)&sAh[m * KP + ak0 + 16]);
            aLo[mi] = cat8(*(const v8bf*)&sAl[m * KP + ak0],
                           *(const v8bf*)&sAl[m * KP + ak0 + 16]);
        }
#pragma unroll
        for (int ni = 0; ni < 2; ++ni) {
            int nn = wn + ni * 16 + am;
            bHi[ni] = cat8(*(const v8bf*)&sBh[nn * KP + bk0],
                           *(const v8bf*)&sBh[nn * KP + bk0 + 16]);
            bLo[ni] = cat8(*(const v8bf*)&sBl[nn * KP + bk0],
                           *(const v8bf*)&sBl[nn * KP + bk0 + 16]);
        }
#pragma unroll
        for (int mi = 0; mi < 2; ++mi)
#pragma unroll
            for (int ni = 0; ni < 2; ++ni) {
                acc[mi][ni] = __builtin_amdgcn_wmma_f32_16x16x32_bf16(
                    false, aHi[mi], false, bHi[ni], (short)0, acc[mi][ni], false, false);
                acc[mi][ni] = __builtin_amdgcn_wmma_f32_16x16x32_bf16(
                    false, aHi[mi], false, bLo[ni], (short)0, acc[mi][ni], false, false);
                acc[mi][ni] = __builtin_amdgcn_wmma_f32_16x16x32_bf16(
                    false, aLo[mi], false, bHi[ni], (short)0, acc[mi][ni], false, false);
            }
    };

    // ---------------- software-pipelined K loop (36 slices, double buffered) -----
    float gA[8];
    v4u   gB[4];
    float gBf[16];

    loadA(0, gA);
    if (PRE) loadB(0, gB); else loadBf(0, gBf);
    storeA(gA, smem);
    if (PRE) storeB(gB, smem); else storeBf(gBf, smem);

    for (int k = 0; k < 36; ++k) {
        __syncthreads();
        unsigned char* cur = smem + (size_t)(k & 1) * BUF_BYTES;
        unsigned char* nxt = smem + (size_t)((k + 1) & 1) * BUF_BYTES;
        if (k + 1 < 36) {                      // issue next-slice global loads early
            loadA(k + 1, gA);
            if (PRE) loadB(k + 1, gB); else loadBf(k + 1, gBf);
        }
        compute(cur);                          // ds_load + wmma on current buffer
        if (k + 1 < 36) {                      // convert + stage into other buffer
            storeA(gA, nxt);
            if (PRE) storeB(gB, nxt); else storeBf(gBf, nxt);
        }
    }

    // ---------------- epilogue: stage f32 via LDS, coalesced NT store + bias -----
    __syncthreads();
    float* sOut = (float*)smem;                // 33280 B, aliases pipeline buffers
#pragma unroll
    for (int mi = 0; mi < 2; ++mi)
#pragma unroll
        for (int ni = 0; ni < 2; ++ni)
#pragma unroll
            for (int g = 0; g < 8; ++g) {
                int m  = wm + mi * 16 + (lane >> 4) * 8 + g;   // C layout: VGPR g -> M
                int nn = wn + ni * 16 + (lane & 15);
                sOut[nn * OPD + m] = acc[mi][ni][g];
            }
    __syncthreads();

    float* outN = out + ((size_t)n_img * 256 + n_base) * 3136 + p_base;
#pragma unroll 4
    for (int i = 0; i < 32; ++i) {
        int idx = i * 256 + t;
        int pl  = idx & 63;                    // consecutive threads -> consecutive pixels
        int nl  = idx >> 6;
        __builtin_nontemporal_store(sOut[nl * OPD + pl] + bias[n_base + nl],
                                    &outN[(size_t)nl * 3136 + pl]);
    }
}

extern "C" void kernel_launch(void* const* d_in, const int* in_sizes, int n_in,
                              void* d_out, int out_size, void* d_ws, size_t ws_size,
                              hipStream_t stream) {
    (void)in_sizes; (void)n_in; (void)out_size;
    const float* x    = (const float*)d_in[0];   // (32,128,56,56)
    const float* wgt  = (const float*)d_in[1];   // (256,128,3,3)
    const float* bias = (const float*)d_in[2];   // (256,)
    float* out = (float*)d_out;                  // (32,256,56,56)

    dim3 grid(32 * 49, 2, 1);
    dim3 block(256, 1, 1);

    const size_t WS_NEED = (size_t)9 * 2 * 256 * 128 * sizeof(__bf16);  // 1.18 MB
    if (d_ws != nullptr && ws_size >= WS_NEED) {
        __bf16* wbf = (__bf16*)d_ws;
        preconv_weights<<<dim3(294912 / 256), block, 0, stream>>>(wgt, wbf);
        conv3x3_igemm_bf16x3<true><<<grid, block, 0, stream>>>(x, wgt, wbf, bias, out);
    } else {
        conv3x3_igemm_bf16x3<false><<<grid, block, 0, stream>>>(x, wgt, nullptr, bias, out);
    }
}
